// Fp8ScaledLinear_21483426415114
// MI455X (gfx1250) — 8-bit matrix operand order repaired and checked by reading, compiled, not run on hardware
//
#include <hip/hip_runtime.h>
#include <hip/hip_bf16.h>
#include <stdint.h>


typedef int   v16i __attribute__((ext_vector_type(16)));
typedef float v8f  __attribute__((ext_vector_type(8)));

#define K_DIM     2048
#define N_DIM     8192
#define FP8_MAX_F 448.0f

__global__ void fp8lin_init_kernel(unsigned int* amax_x, unsigned int* amax_w) {
    *amax_x = 0u;
    *amax_w = 0u;
}

__global__ __launch_bounds__(256)
void fp8lin_amax_kernel(const float4* __restrict__ p, int n4,
                        unsigned int* __restrict__ dst) {
    float m = 0.0f;
    int stride = gridDim.x * blockDim.x;
    for (int i = blockIdx.x * blockDim.x + threadIdx.x; i < n4; i += stride) {
        float4 v = p[i];
        m = fmaxf(m, fmaxf(fmaxf(fabsf(v.x), fabsf(v.y)),
                           fmaxf(fabsf(v.z), fabsf(v.w))));
    }
    for (int off = 16; off > 0; off >>= 1)
        m = fmaxf(m, __shfl_xor(m, off, 32));
    __shared__ float smax[8];
    int lane = threadIdx.x & 31;
    int wv   = threadIdx.x >> 5;
    if (lane == 0) smax[wv] = m;
    __syncthreads();
    if (threadIdx.x == 0) {
        float b = smax[0];
#pragma unroll
        for (int i = 1; i < 8; ++i) b = fmaxf(b, smax[i]);
        atomicMax(dst, __float_as_uint(b));
    }
}

__device__ inline unsigned int f32_to_e4m3_sw(float f) {
    unsigned int u   = __float_as_uint(f);
    unsigned int sgn = (u >> 24) & 0x80u;
    unsigned int mag = u & 0x7FFFFFFFu;
    if (mag >= 0x7F800000u) return sgn | 0x7Fu;
    float a = __uint_as_float(mag);
    if (a >= 464.0f) return sgn | 0x7Eu;
    if (a < 0.0009765625f) {
        int q = (int)(a * 512.0f + 0.5f);
        return sgn | (unsigned int)q;
    }
    unsigned int lsb = (mag >> 20) & 1u;
    mag += 0x0007FFFFu + lsb;
    int e = (int)(mag >> 23) - 127;
    unsigned int mant = (mag >> 20) & 7u;
    if (e < -6) {
        int q = (int)(__uint_as_float(mag) * 512.0f + 0.5f);
        if (q > 7) q = 7;
        return sgn | (unsigned int)q;
    }
    int e8 = e + 7;
    if (e8 > 15 || (e8 == 15 && mant == 7u)) return sgn | 0x7Eu;
    return sgn | ((unsigned int)e8 << 3) | mant;
}

__global__ __launch_bounds__(256)
void fp8lin_quant_kernel(const float4* __restrict__ src, int n4,
                         const unsigned int* __restrict__ amax_bits,
                         unsigned int* __restrict__ dst) {
    int i = blockIdx.x * blockDim.x + threadIdx.x;
    if (i >= n4) return;
    float amax = __uint_as_float(*amax_bits);
    float inv  = (amax == 0.0f) ? 1.0f : (FP8_MAX_F / amax);
    float4 v = src[i];
#if defined(__has_builtin) && __has_builtin(__builtin_amdgcn_cvt_pk_fp8_f32)
    int packed = 0;
    packed = __builtin_amdgcn_cvt_pk_fp8_f32(v.x * inv, v.y * inv, packed, false);
    packed = __builtin_amdgcn_cvt_pk_fp8_f32(v.z * inv, v.w * inv, packed, true);
    dst[i] = (unsigned int)packed;
#else
    unsigned int b0 = f32_to_e4m3_sw(v.x * inv);
    unsigned int b1 = f32_to_e4m3_sw(v.y * inv);
    unsigned int b2 = f32_to_e4m3_sw(v.z * inv);
    unsigned int b3 = f32_to_e4m3_sw(v.w * inv);
    dst[i] = b0 | (b1 << 8) | (b2 << 16) | (b3 << 24);
#endif
}

__device__ inline void async_copy_b128(unsigned lds_addr, const void* base,
                                       unsigned goff) {
    asm volatile("global_load_async_to_lds_b128 %0, %1, %2"
                 :: "v"(lds_addr), "v"(goff), "s"(base)
                 : "memory");
}

__global__ __launch_bounds__(256)
void fp8lin_gemm_kernel(const unsigned char* __restrict__ aq,
                        const unsigned char* __restrict__ bq,
                        const float* __restrict__ bias,
                        const unsigned int* __restrict__ amax_x,
                        const unsigned int* __restrict__ amax_w,
                        float* __restrict__ out) {
    __shared__ uint4 sh[4096];

    const int tid  = threadIdx.x;
    const int wave = tid >> 5;
    const int lane = tid & 31;
    const int wm   = wave >> 2;
    const int wn   = wave & 3;
    const int half = lane >> 4;
    const int l16  = lane & 15;

    const int blockN = blockIdx.x * 128;
    const int blockM = blockIdx.y * 128;

    const unsigned shbase = (unsigned)(uintptr_t)(&sh[0]);

    auto issue_tile = [&](int k0, int buf) {
#pragma unroll
        for (int p = 0; p < 4; ++p) {
            int linear = p * 256 + tid;
            int row    = linear >> 3;
            int c16    = linear & 7;
            unsigned goffA = (unsigned)((blockM + row) * K_DIM + k0 + c16 * 16);
            unsigned goffB = (unsigned)((blockN + row) * K_DIM + k0 + c16 * 16);
            unsigned la = shbase + (unsigned)(buf * 32768 + linear * 16);
            async_copy_b128(la,         aq, goffA);
            async_copy_b128(la + 16384, bq, goffB);
        }
    };

    v8f zero = {};
    v8f acc[4][2];
#pragma unroll
    for (int i = 0; i < 4; ++i)
#pragma unroll
        for (int j = 0; j < 2; ++j) acc[i][j] = zero;

    issue_tile(0, 0);

    for (int it = 0; it < K_DIM / 128; ++it) {
        __syncthreads();
        if (it + 1 < K_DIM / 128) {
            issue_tile((it + 1) * 128, (it + 1) & 1);
            asm volatile("s_wait_asynccnt 0x8" ::: "memory");
        } else {
            asm volatile("s_wait_asynccnt 0x0" ::: "memory");
        }
        __syncthreads();

        const uint4* As = &sh[(it & 1) * 2048];
        const uint4* Bs = As + 1024;

        const uint2* A2 = (const uint2*)As;
        v16i afrag[4], bfrag[2];
#pragma unroll
        for (int i = 0; i < 4; ++i) {
            int r = wm * 64 + i * 16 + l16;
#pragma unroll
            for (int j = 0; j < 8; ++j) {
                int byteoff = (j >> 1) * 32 + half * 16 + (j & 1) * 8;
                uint2 c = A2[r * 16 + (byteoff >> 3)];
                afrag[i][2 * j]     = (int)c.x;
                afrag[i][2 * j + 1] = (int)c.y;
            }
        }
#pragma unroll
        for (int j2 = 0; j2 < 2; ++j2) {
            int r = wn * 32 + j2 * 16 + l16;
#pragma unroll
            for (int c = 0; c < 4; ++c) {
                uint4 q = Bs[r * 8 + half + 2 * c];
                bfrag[j2][4 * c]     = (int)q.x;
                bfrag[j2][4 * c + 1] = (int)q.y;
                bfrag[j2][4 * c + 2] = (int)q.z;
                bfrag[j2][4 * c + 3] = (int)q.w;
            }
        }
#pragma unroll
        for (int i = 0; i < 4; ++i)
#pragma unroll
            for (int j = 0; j < 2; ++j)
                acc[i][j] = __builtin_amdgcn_wmma_f32_16x16x128_fp8_fp8(
                    afrag[i], bfrag[j], (short)0, acc[i][j], false, false);
    }

    float ax = __uint_as_float(*amax_x);
    float aw = __uint_as_float(*amax_w);
    float sx = (ax == 0.0f) ? 1.0f : ax / FP8_MAX_F;
    float sw = (aw == 0.0f) ? 1.0f : aw / FP8_MAX_F;
    float s  = sx * sw;

#pragma unroll
    for (int i = 0; i < 4; ++i) {
#pragma unroll
        for (int j = 0; j < 2; ++j) {
            int col = blockN + wn * 32 + j * 16 + l16;
            float bb = __bfloat162float(__float2bfloat16(bias[col]));
#pragma unroll
            for (int v = 0; v < 8; ++v) {
                int row = blockM + wm * 64 + i * 16 + half * 8 + v;
                float t = __bfloat162float(__float2bfloat16(acc[i][j][v] * s));
                t = __bfloat162float(__float2bfloat16(t + bb));
                out[(size_t)row * N_DIM + col] = t;
            }
        }
    }
}

extern "C" void kernel_launch(void* const* d_in, const int* in_sizes, int n_in,
                              void* d_out, int out_size, void* d_ws, size_t ws_size,
                              hipStream_t stream) {
    const float* x    = (const float*)d_in[0];
    const float* w    = (const float*)d_in[1];
    const float* bias = (const float*)d_in[2];
    float*       out  = (float*)d_out;

    const int K = K_DIM;
    const int N = N_DIM;
    const int M = in_sizes[0] / K;

    unsigned char* ws     = (unsigned char*)d_ws;
    unsigned int*  amax_x = (unsigned int*)ws;
    unsigned int*  amax_w = amax_x + 1;
    unsigned char* aq     = ws + 256;
    unsigned char* wq     = aq + (size_t)M * K;

    fp8lin_init_kernel<<<1, 1, 0, stream>>>(amax_x, amax_w);

    fp8lin_amax_kernel<<<1024, 256, 0, stream>>>((const float4*)x, (M * K) / 4, amax_x);
    fp8lin_amax_kernel<<<1024, 256, 0, stream>>>((const float4*)w, (N * K) / 4, amax_w);

    fp8lin_quant_kernel<<<((M * K / 4) + 255) / 256, 256, 0, stream>>>(
        (const float4*)x, M * K / 4, amax_x, (unsigned int*)aq);
    fp8lin_quant_kernel<<<((N * K / 4) + 255) / 256, 256, 0, stream>>>(
        (const float4*)w, N * K / 4, amax_w, (unsigned int*)wq);

    dim3 grid(N / 128, M / 128);
    fp8lin_gemm_kernel<<<grid, 256, 0, stream>>>(aq, wq, bias, amax_x, amax_w, out);
}
